// SSM_51264729645090
// MI455X (gfx1250) — matrix operand order repaired and checked by reading, not compiled, not run on hardware
//
#include <hip/hip_runtime.h>
#include <hip/hip_bf16.h>

#define BATCH 8
#define SEQ   1024
#define DIM   128
#define CHUNK 64
#define NCHUNK (SEQ/CHUNK)        // 16 chunks per batch
#define NBC    (BATCH*NCHUNK)     // 128 chunk-blocks total
#define LDSTR  132                // padded bf16 row stride (bank-conflict avoidance)

typedef __attribute__((ext_vector_type(16))) __bf16 v16bf;
typedef __attribute__((ext_vector_type(8)))  float  v8f;

__device__ __forceinline__ unsigned short f32_bf16(float f) {
  unsigned u = __builtin_bit_cast(unsigned, f);
  u += 0x7FFFu + ((u >> 16) & 1u);      // round-to-nearest-even
  return (unsigned short)(u >> 16);
}

// A-fragment: 16x32 bf16, ISA layout: lane<16 holds row M=lane, K in {0..7,16..23};
// lane>=16 holds row M=lane-16, K in {8..15,24..31}; pairs per VGPR.
__device__ __forceinline__ v16bf load_a_frag(const unsigned short* Alds, int mtile, int kt, int lane) {
  const int row  = mtile * 16 + (lane & 15);
  const int half = lane >> 4;
  const unsigned* p = (const unsigned*)(Alds + row * LDSTR + kt * 32 + half * 8);
  union { unsigned u[8]; v16bf v; } r;
  r.u[0] = p[0]; r.u[1] = p[1]; r.u[2] = p[2];  r.u[3] = p[3];
  r.u[4] = p[8]; r.u[5] = p[9]; r.u[6] = p[10]; r.u[7] = p[11];
  return r.v;
}

// B-fragment: 32x16 bf16, lanes 0-15 hold K in {0..7,16..23} (N=lane), lanes 16-31 hold K in {8..15,24..31}.
// Mold is stored TRANSPOSED in LDS (MT[n][k]) so each lane reads two runs of 8 contiguous bf16.
__device__ __forceinline__ v16bf load_b_frag(const unsigned short* MT, int ntile, int kt, int lane) {
  const int col  = ntile * 16 + (lane & 15);
  const int half = lane >> 4;
  const unsigned* p = (const unsigned*)(MT + col * LDSTR + kt * 32 + half * 8);
  union { unsigned u[8]; v16bf v; } r;
#pragma unroll
  for (int i = 0; i < 8; ++i) r.u[i] = p[i < 4 ? i : i + 4];
  return r.v;
}

// ---------- Pass 0: Bx[b,s,:] = B[b,s] @ x[b,s] (streams B once) ----------
__global__ void bx_kernel(const float* __restrict__ Bm, const float* __restrict__ x,
                          float* __restrict__ Bx) {
  const int bs = blockIdx.x;          // 0..8191
  const int i  = threadIdx.x;         // 0..127
  __shared__ float xs[DIM];
  xs[i] = x[(size_t)bs * DIM + i];
  __syncthreads();
  const float4* row = (const float4*)(Bm + (size_t)bs * DIM * DIM + (size_t)i * DIM);
  float acc = 0.f;
#pragma unroll 8
  for (int j4 = 0; j4 < DIM / 4; ++j4) {
    float4 a = row[j4];
    acc += a.x * xs[4*j4] + a.y * xs[4*j4+1] + a.z * xs[4*j4+2] + a.w * xs[4*j4+3];
  }
  Bx[(size_t)bs * DIM + i] = acc;
}

// ---------- Pass 1: per chunk, M_c = prod A_t (bf16 WMMA), v_c = chunk scan from 0 (fp32) ----------
__global__ void phase1_kernel(const float* __restrict__ A, const float* __restrict__ Bx,
                              float* __restrict__ Mc, float* __restrict__ vc) {
  extern __shared__ char smem[];
  unsigned short* A_lds = (unsigned short*)smem;                      // 128*132*2 = 33792 B
  unsigned short* Mb0   = (unsigned short*)(smem + 33792);            // M^T double buffer
  unsigned short* Mb1   = (unsigned short*)(smem + 67584);
  float*          v_lds = (float*)(smem + 101376);                    // 512 B

  const int tid = threadIdx.x, lane = tid & 31, wave = tid >> 5;      // 256 threads, 8 waves
  const int bc = blockIdx.x;
  const int b = bc / NCHUNK, c = bc % NCHUNK;
  const int s0 = c * CHUNK;

  // M^T = Identity, v = 0
  for (int idx = tid; idx < DIM * LDSTR; idx += blockDim.x) {
    const int r = idx / LDSTR, k = idx % LDSTR;
    Mb0[idx] = (r == k) ? (unsigned short)0x3F80 : (unsigned short)0;
  }
  if (tid < DIM) v_lds[tid] = 0.f;
  __syncthreads();

  const int row = tid >> 1;            // 2 threads per matrix row
  const int jb  = (tid & 1) * 64;

  for (int t = 0; t < CHUNK; ++t) {
    const int s = s0 + t;
    const float* Ag = A + (size_t)(b * SEQ + s) * DIM * DIM;
    if (t + 1 < CHUNK) {               // prefetch next timestep's 64 KB A matrix
      const float* An = Ag + DIM * DIM;
      __builtin_prefetch(An + tid * 64, 0, 1);
      __builtin_prefetch(An + tid * 64 + 32, 0, 1);
    }
    // Stream A_t: fp32 matvec partial (v update) fused with bf16 conversion into LDS
    float p = 0.f;
    const float4* arow = (const float4*)(Ag + (size_t)row * DIM + jb);
#pragma unroll
    for (int jj = 0; jj < 16; ++jj) {
      float4 a = arow[jj];
      const int j = jb + jj * 4;
      p += a.x * v_lds[j] + a.y * v_lds[j+1] + a.z * v_lds[j+2] + a.w * v_lds[j+3];
      unsigned lo = (unsigned)f32_bf16(a.x) | ((unsigned)f32_bf16(a.y) << 16);
      unsigned hi = (unsigned)f32_bf16(a.z) | ((unsigned)f32_bf16(a.w) << 16);
      unsigned* dst = (unsigned*)(A_lds + row * LDSTR + j);
      dst[0] = lo; dst[1] = hi;
    }
    p += __shfl_xor(p, 1, 32);
    __syncthreads();                   // v fully read; A_lds complete
    if (!(tid & 1))
      v_lds[row] = p + Bx[(size_t)(b * SEQ + s) * DIM + row];

    // Mnew^T = (A_t * Mold)^T via bf16 WMMA, fp32 accumulate. Wave w owns mtile=w.
    const unsigned short* Mold = (t & 1) ? Mb1 : Mb0;
    unsigned short*       Mnew = (t & 1) ? Mb0 : Mb1;
    const int mtile = wave;
    const int half  = lane >> 4;
    v16bf af[4];
#pragma unroll
    for (int kt = 0; kt < 4; ++kt) af[kt] = load_a_frag(A_lds, mtile, kt, lane);
#pragma unroll
    for (int nt = 0; nt < 8; ++nt) {
      v8f acc = {0.f, 0.f, 0.f, 0.f, 0.f, 0.f, 0.f, 0.f};
#pragma unroll
      for (int kt = 0; kt < 4; ++kt) {
        v16bf bf = load_b_frag(Mold, nt, kt, lane);
        acc = __builtin_amdgcn_wmma_f32_16x16x32_bf16(false, af[kt], false, bf,
                                                      (short)0, acc, false, false);
      }
      const int ng = nt * 16 + (lane & 15);
      if (t == CHUNK - 1) {            // final product -> Mc in fp32 (row-major)
        float* McB = Mc + (size_t)bc * DIM * DIM;
#pragma unroll
        for (int r = 0; r < 8; ++r) {
          const int mg = mtile * 16 + r + half * 8;   // f32 C/D layout
          McB[(size_t)mg * DIM + ng] = acc[r];
        }
      } else {                          // intermediate -> bf16 transposed buffer
#pragma unroll
        for (int r = 0; r < 8; ++r) {
          const int mg = mtile * 16 + r + half * 8;
          Mnew[ng * LDSTR + mg] = f32_bf16(acc[r]);
        }
      }
    }
    __syncthreads();
  }
  if (tid < DIM) vc[(size_t)bc * DIM + tid] = v_lds[tid];
}

// ---------- Pass 2: sequential chunk combine per batch: h_{c+1} = M_c h_c + v_c ----------
__global__ void combine_kernel(const float* __restrict__ Mc, const float* __restrict__ vc,
                               float* __restrict__ hinit) {
  __shared__ float h[DIM];
  const int b = blockIdx.x, i = threadIdx.x;   // 8 blocks x 128 threads
  h[i] = 0.f;
  __syncthreads();
  for (int c = 0; c < NCHUNK; ++c) {
    const int bc = b * NCHUNK + c;
    hinit[(size_t)bc * DIM + i] = h[i];
    const float* Mrow = Mc + (size_t)bc * DIM * DIM + (size_t)i * DIM;
    float acc = vc[(size_t)bc * DIM + i];
#pragma unroll 8
    for (int j = 0; j < DIM; ++j) acc += Mrow[j] * h[j];
    __syncthreads();
    h[i] = acc;
    __syncthreads();
  }
}

// ---------- Pass 3: exact fp32 re-scan per chunk from hinit, emit y = C .* h ----------
__global__ void phase3_kernel(const float* __restrict__ A, const float* __restrict__ Bx,
                              const float* __restrict__ C, const float* __restrict__ hinit,
                              float* __restrict__ y) {
  __shared__ float h[DIM];
  const int tid = threadIdx.x;                 // 256 threads
  const int bc = blockIdx.x;
  const int b = bc / NCHUNK, c = bc % NCHUNK;
  if (tid < DIM) h[tid] = hinit[(size_t)bc * DIM + tid];
  __syncthreads();
  const int row = tid >> 1;
  const int jb  = (tid & 1) * 64;
  for (int t = 0; t < CHUNK; ++t) {
    const int s = c * CHUNK + t;
    const float* Ag = A + (size_t)(b * SEQ + s) * DIM * DIM;
    if (t + 1 < CHUNK) {
      const float* An = Ag + DIM * DIM;
      __builtin_prefetch(An + tid * 64, 0, 1);
      __builtin_prefetch(An + tid * 64 + 32, 0, 1);
    }
    float p = 0.f;
    const float4* arow = (const float4*)(Ag + (size_t)row * DIM + jb);
#pragma unroll
    for (int jj = 0; jj < 16; ++jj) {
      float4 a = arow[jj];
      const int j = jb + jj * 4;
      p += a.x * h[j] + a.y * h[j+1] + a.z * h[j+2] + a.w * h[j+3];
    }
    p += __shfl_xor(p, 1, 32);
    __syncthreads();                 // all reads of old h done
    if (!(tid & 1)) {
      const size_t off = (size_t)(b * SEQ + s) * DIM + row;
      const float hn = p + Bx[off];
      h[row] = hn;
      y[off] = C[off] * hn;
    }
    __syncthreads();
  }
}

extern "C" void kernel_launch(void* const* d_in, const int* in_sizes, int n_in,
                              void* d_out, int out_size, void* d_ws, size_t ws_size,
                              hipStream_t stream) {
  const float* A  = (const float*)d_in[0];
  const float* Bm = (const float*)d_in[1];
  const float* C  = (const float*)d_in[2];
  const float* x  = (const float*)d_in[3];
  float* out = (float*)d_out;

  float* ws    = (float*)d_ws;
  float* Bx    = ws;                               // 8*1024*128      = 1,048,576 f32
  float* Mc    = Bx + (size_t)BATCH * SEQ * DIM;   // 128*128*128     = 2,097,152 f32
  float* vc    = Mc + (size_t)NBC * DIM * DIM;     // 128*128         =    16,384 f32
  float* hinit = vc + (size_t)NBC * DIM;           // 128*128         =    16,384 f32
                                                   // total ~12.7 MB of workspace

  bx_kernel<<<BATCH * SEQ, DIM, 0, stream>>>(Bm, x, Bx);
  phase1_kernel<<<NBC, 256, 101888, stream>>>(A, Bx, Mc, vc);
  combine_kernel<<<BATCH, DIM, 0, stream>>>(Mc, vc, hinit);
  phase3_kernel<<<NBC, 256, 0, stream>>>(A, Bx, C, hinit, out);
}
